// PointMAE2DLiDARDetector_60541859004648
// MI455X (gfx1250) — matrix operand order repaired and checked by reading, not compiled, not run on hardware
//
#include <hip/hip_runtime.h>
#include <math.h>

// ---------------- problem constants ----------------
#define BB   16
#define NN   8192
#define GG   256
#define KK   64
#define DD   256
#define TT   257          // G+1 tokens
#define ROWS (BB*TT)      // 4112
#define HH   8
#define DH   32
#define HID  1024
#define PH   128
#define BN_SCALE 0.9999950000374997f

typedef __attribute__((ext_vector_type(16))) _Float16 v16h;
typedef __attribute__((ext_vector_type(8)))  float    v8f;
typedef unsigned int u32x4 __attribute__((ext_vector_type(4)));
typedef int          i32x4 __attribute__((ext_vector_type(4)));
typedef int          i32x8 __attribute__((ext_vector_type(8)));

#if __has_builtin(__builtin_amdgcn_tensor_load_to_lds) && __has_builtin(__builtin_amdgcn_s_wait_tensorcnt)
#define HAVE_TDM 1
#else
#define HAVE_TDM 0
#endif

// ---------------- WMMA helpers (CDNA5 16x16x32 f16) ----------------
__device__ __forceinline__ v8f wmma16(v16h a, v16h b, v8f c) {
    return __builtin_amdgcn_wmma_f32_16x16x32_f16(false, a, false, b, (short)0, c, false, false);
}

// A fragment (16x32 f16) from LDS, row-major lda.
// Lane l<16: row=l, K=0..15; lane>=16: row=l-16, K=16..31
__device__ __forceinline__ v16h load_a_lds(const _Float16* sA, int lda, int m0, int k0, int lane) {
    const _Float16* p = sA + (m0 + (lane & 15)) * lda + k0 + ((lane >= 16) ? 16 : 0);
    v16h a;
#pragma unroll
    for (int i = 0; i < 8; i++) a[i] = p[i];
#pragma unroll
    for (int i = 0; i < 8; i++) a[8 + i] = p[8 + i];
    return a;
}

// B fragment (32x16 f16) from global, W is KxN row-major (ldw=N).
// Lane l<16: col=l, K=0..15; lane>=16: col=l-16, K=16..31
__device__ __forceinline__ v16h load_b_glb(const _Float16* W, int ldw, int k0, int n0, int lane) {
    int col = n0 + (lane & 15);
    int kb  = k0 + ((lane >= 16) ? 16 : 0);
    v16h b;
#pragma unroll
    for (int i = 0; i < 16; i++) b[i] = W[(kb + i) * ldw + col];
    return b;
}

// B fragment (32x16 f16) from an LDS slab (32 rows x 128 cols, row-major)
__device__ __forceinline__ v16h load_b_lds(const _Float16* sB, int n0, int lane) {
    const _Float16* p = sB + ((lane >= 16) ? 16 : 0) * 128 + n0 + (lane & 15);
    v16h b;
#pragma unroll
    for (int i = 0; i < 16; i++) b[i] = p[i * 128];
    return b;
}

// ---------------- TDM: DMA one 32x128 f16 weight slab into LDS ----------------
// Descriptor layout per CDNA5 ISA (2-D tile, groups 2/3 zero).
#if HAVE_TDM
__device__ __forceinline__ void tdm_load_b_slab(const _Float16* W, int N, int K, int k0, int nblk,
                                                _Float16* dst) {
    unsigned long long ga = (unsigned long long)(const void*)(W + (size_t)k0 * N + nblk);
    unsigned lds_off = (unsigned)(unsigned long long)(void*)dst; // low 32b of flat LDS addr = LDS offset
    u32x4 g0;
    g0[0] = 1u;                                            // count=1, user descriptor
    g0[1] = lds_off;                                       // lds_addr
    g0[2] = (unsigned)(ga & 0xffffffffu);                  // global_addr[31:0]
    g0[3] = (unsigned)((ga >> 32) & 0x1ffffffu) | (2u << 30); // global_addr[56:32] | type=2
    unsigned td0 = (unsigned)N;        // tensor_dim0 (elements per row)
    unsigned td1 = (unsigned)K;        // tensor_dim1 (rows)
    unsigned tile0 = 128u;             // tile cols
    unsigned tile1 = 32u;              // tile rows
    unsigned long long s0 = (unsigned long long)N;  // tensor_dim0_stride (elements)
    i32x8 g1;
    g1[0] = (int)(1u << 16);                                   // wg_mask=0, data_size=1 (2B)
    g1[1] = (int)((td0 & 0xffffu) << 16);                      // tensor_dim0[15:0]
    g1[2] = (int)((td0 >> 16) | ((td1 & 0xffffu) << 16));      // tensor_dim0[31:16] | tensor_dim1[15:0]
    g1[3] = (int)((td1 >> 16) | (tile0 << 16));                // tensor_dim1[31:16] | tile_dim0
    g1[4] = (int)(tile1 & 0xffffu);                            // tile_dim1, tile_dim2=0
    g1[5] = (int)(unsigned)(s0 & 0xffffffffu);                 // tensor_dim0_stride[31:0]
    g1[6] = (int)(unsigned)((s0 >> 32) & 0xffffu);             // stride[47:32] | dim1_stride[15:0]=0
    g1[7] = 0;
    i32x4 gz = {0, 0, 0, 0};
#if defined(__clang_major__) && (__clang_major__ >= 23)
    i32x8 gz8 = {0, 0, 0, 0, 0, 0, 0, 0};
    __builtin_amdgcn_tensor_load_to_lds(g0, g1, gz, gz, gz8, 0);
#else
    __builtin_amdgcn_tensor_load_to_lds(g0, g1, gz, gz, 0);
#endif
}
#endif

// stage slab rows [k0,k0+32) x cols [nblk,nblk+128) of W(KxN) into dst (LDS)
__device__ __forceinline__ void stage_b_slab(const _Float16* W, int N, int K, int k0, int nblk,
                                             _Float16* dst, int wave, int tid) {
#if HAVE_TDM
    if (wave == 0) tdm_load_b_slab(W, N, K, k0, nblk, dst);
#else
    for (int i = tid; i < 32 * 128; i += 256) {
        int r = i >> 7, c = i & 127;
        dst[i] = W[(size_t)(k0 + r) * N + nblk + c];
    }
#endif
}

__device__ __forceinline__ void stage_b_wait(int wave) {
#if HAVE_TDM
    if (wave == 0) __builtin_amdgcn_s_wait_tensorcnt(0);
#else
    (void)wave;
#endif
}

// ---------------- weight conversion ----------------
__global__ void f32_to_f16_kernel(const float* __restrict__ s, _Float16* __restrict__ d, int n) {
    int i = blockIdx.x * 256 + threadIdx.x;
    if (i < n) d[i] = (_Float16)s[i];
}

__global__ void zero_kernel(float* __restrict__ p, int n) {
    int i = blockIdx.x * 256 + threadIdx.x;
    if (i < n) p[i] = 0.f;
}

// ---------------- FPS: one block per batch ----------------
__global__ void fps_kernel(const float* __restrict__ x, float* __restrict__ centers,
                           int* __restrict__ cidx) {
    int b = blockIdx.x;
    const float* xb = x + (size_t)b * NN * 3;
    __shared__ float dist[NN];
    __shared__ float rmax[256];
    __shared__ int   rarg[256];
    __shared__ float cen[3];
    __shared__ int   curFar;
    int tid = threadIdx.x;
    for (int i = tid; i < NN; i += 256) dist[i] = 1e10f;
    if (tid == 0) curFar = 0;
    __syncthreads();
    for (int g = 0; g < GG; g++) {
        int far = curFar;
        if (tid == 0) {
            cidx[b * GG + g] = far;
            cen[0] = xb[far * 3 + 0];
            cen[1] = xb[far * 3 + 1];
            cen[2] = xb[far * 3 + 2];
            centers[(b * GG + g) * 3 + 0] = cen[0];
            centers[(b * GG + g) * 3 + 1] = cen[1];
            centers[(b * GG + g) * 3 + 2] = cen[2];
        }
        __syncthreads();
        float bm = -1.f; int ba = 0;
        for (int i = tid; i < NN; i += 256) {
            float dx = xb[i * 3 + 0] - cen[0];
            float dy = xb[i * 3 + 1] - cen[1];
            float dz = xb[i * 3 + 2] - cen[2];
            float d  = dx * dx + dy * dy + dz * dz;
            float od = dist[i];
            if (d < od) { od = d; dist[i] = d; }
            if (od > bm) { bm = od; ba = i; }
        }
        rmax[tid] = bm; rarg[tid] = ba;
        __syncthreads();
        for (int s = 128; s > 0; s >>= 1) {
            if (tid < s && rmax[tid + s] > rmax[tid]) { rmax[tid] = rmax[tid + s]; rarg[tid] = rarg[tid + s]; }
            __syncthreads();
        }
        if (tid == 0) curFar = rarg[0];
        __syncthreads();
    }
}

// ---------------- KNN: one block per (b,g) ----------------
__global__ void knn_kernel(const float* __restrict__ x, const float* __restrict__ centers,
                           int* __restrict__ idx, float* __restrict__ neigh) {
    int bg = blockIdx.x;
    int b  = bg >> 8;
    const float* xb = x + (size_t)b * NN * 3;
    __shared__ float d2[NN];
    __shared__ float rmin[256];
    __shared__ int   rarg[256];
    float cx = centers[bg * 3 + 0], cy = centers[bg * 3 + 1], cz = centers[bg * 3 + 2];
    int tid = threadIdx.x;
    for (int i = tid; i < NN; i += 256) {
        float dx = xb[i * 3 + 0] - cx, dy = xb[i * 3 + 1] - cy, dz = xb[i * 3 + 2] - cz;
        d2[i] = dx * dx + dy * dy + dz * dz;
    }
    __syncthreads();
    for (int k = 0; k < KK; k++) {
        float bm = 3.4e38f; int ba = 0;
        for (int i = tid; i < NN; i += 256) {
            float d = d2[i];
            if (d < bm) { bm = d; ba = i; }
        }
        rmin[tid] = bm; rarg[tid] = ba;
        __syncthreads();
        for (int s = 128; s > 0; s >>= 1) {
            if (tid < s && rmin[tid + s] < rmin[tid]) { rmin[tid] = rmin[tid + s]; rarg[tid] = rarg[tid + s]; }
            __syncthreads();
        }
        if (tid == 0) {
            int j = rarg[0];
            idx[bg * KK + k] = j;
            neigh[(bg * KK + k) * 3 + 0] = xb[j * 3 + 0] - cx;
            neigh[(bg * KK + k) * 3 + 1] = xb[j * 3 + 1] - cy;
            neigh[(bg * KK + k) * 3 + 2] = xb[j * 3 + 2] - cz;
            d2[j] = 3.4e38f;
        }
        __syncthreads();
    }
}

// ---------------- fused group encoder: one block per group, all GEMMs in LDS ----------------
// dyn LDS: sa = 64x512 f16 (64KB), sc = 64x512 f16 / 64x256 f32 (64KB)
__global__ void encoder_kernel(const float* __restrict__ neigh,
                               const float* __restrict__ w1, const float* __restrict__ b1,
                               const float* __restrict__ g1, const float* __restrict__ be1,
                               const _Float16* __restrict__ W2, const float* __restrict__ b2,
                               const _Float16* __restrict__ W3, const float* __restrict__ b3,
                               const float* __restrict__ g2, const float* __restrict__ be2,
                               const _Float16* __restrict__ W4, const float* __restrict__ b4,
                               float* __restrict__ tokens) {
    extern __shared__ char smem[];
    _Float16* sa = (_Float16*)smem;             // 64 x 512
    _Float16* sc = (_Float16*)(smem + 65536);   // 64 x 512
    float*    scf = (float*)sc;                 // alias: 64 x 256 f32
    int bg = blockIdx.x;
    const float* pg = neigh + (size_t)bg * KK * 3;
    int tid = threadIdx.x;
    int wave = tid >> 5, lane = tid & 31;

    // stage 1: h1 = relu(bn1(pg @ w1 + b1))  -> sa (64x128, lda 128)
    for (int i = tid; i < 64 * 128; i += 256) {
        int r = i >> 7, c = i & 127;
        float s = b1[c];
#pragma unroll
        for (int t = 0; t < 3; t++) s += pg[r * 3 + t] * w1[t * 128 + c];
        s = s * g1[c] * BN_SCALE + be1[c];
        sa[r * 128 + c] = (_Float16)fmaxf(s, 0.f);
    }
    __syncthreads();

    // GEMM1: h2 = h1(64x128) @ W2(128x256) + b2 -> sc[:,256..511] (lda 512)
    for (int t = wave; t < 4 * 16; t += 8) {
        int mt = t >> 4, nt = t & 15;
        v8f acc = {0.f,0.f,0.f,0.f,0.f,0.f,0.f,0.f};
        for (int k0 = 0; k0 < 128; k0 += 32) {
            __builtin_prefetch(W2 + (size_t)(k0 + 32) * 256 + nt * 16, 0, 0);
            v16h a = load_a_lds(sa, 128, mt * 16, k0, lane);
            v16h b = load_b_glb(W2, 256, k0, nt * 16, lane);
            acc = wmma16(a, b, acc);
        }
        int col = nt * 16 + (lane & 15);
        int rb  = mt * 16 + ((lane >= 16) ? 8 : 0);
        float bv = b2[col];
#pragma unroll
        for (int v = 0; v < 8; v++) sc[(rb + v) * 512 + 256 + col] = (_Float16)(acc[v] + bv);
    }
    __syncthreads();

    // gmax over rows, broadcast into sc[:,0..255]
    {
        int c = tid; // 256 threads, one column each
        float m = (float)sc[256 + c];
        for (int r = 1; r < 64; r++) { float v = (float)sc[r * 512 + 256 + c]; m = fmaxf(m, v); }
        _Float16 mh = (_Float16)m;
        for (int r = 0; r < 64; r++) sc[r * 512 + c] = mh;
    }
    __syncthreads();

    // GEMM2: h3 = relu(bn2(cat(64x512) @ W3(512x512) + b3)) -> sa (64x512)
    for (int t = wave; t < 4 * 32; t += 8) {
        int mt = t >> 5, nt = t & 31;
        v8f acc = {0.f,0.f,0.f,0.f,0.f,0.f,0.f,0.f};
        for (int k0 = 0; k0 < 512; k0 += 32) {
            __builtin_prefetch(W3 + (size_t)(k0 + 32) * 512 + nt * 16, 0, 0);
            v16h a = load_a_lds(sc, 512, mt * 16, k0, lane);
            v16h b = load_b_glb(W3, 512, k0, nt * 16, lane);
            acc = wmma16(a, b, acc);
        }
        int col = nt * 16 + (lane & 15);
        int rb  = mt * 16 + ((lane >= 16) ? 8 : 0);
        float bv = b3[col], gg = g2[col] * BN_SCALE, bb = be2[col];
#pragma unroll
        for (int v = 0; v < 8; v++) {
            float val = (acc[v] + bv) * gg + bb;
            sa[(rb + v) * 512 + col] = (_Float16)fmaxf(val, 0.f);
        }
    }
    __syncthreads();

    // GEMM3: h4 = h3(64x512) @ W4(512x256) + b4 -> scf (64x256 f32)
    for (int t = wave; t < 4 * 16; t += 8) {
        int mt = t >> 4, nt = t & 15;
        v8f acc = {0.f,0.f,0.f,0.f,0.f,0.f,0.f,0.f};
        for (int k0 = 0; k0 < 512; k0 += 32) {
            __builtin_prefetch(W4 + (size_t)(k0 + 32) * 256 + nt * 16, 0, 0);
            v16h a = load_a_lds(sa, 512, mt * 16, k0, lane);
            v16h b = load_b_glb(W4, 256, k0, nt * 16, lane);
            acc = wmma16(a, b, acc);
        }
        int col = nt * 16 + (lane & 15);
        int rb  = mt * 16 + ((lane >= 16) ? 8 : 0);
        float bv = b4[col];
#pragma unroll
        for (int v = 0; v < 8; v++) scf[(rb + v) * 256 + col] = acc[v] + bv;
    }
    __syncthreads();

    // token = max over rows
    {
        int c = tid;
        float m = scf[c];
        for (int r = 1; r < 64; r++) m = fmaxf(m, scf[r * 256 + c]);
        tokens[(size_t)bg * 256 + c] = m;
    }
}

// ---------------- positional embedding: one block per (b,g) ----------------
__global__ void pos_kernel(const float* __restrict__ centers,
                           const float* __restrict__ w1, const float* __restrict__ b1,
                           const float* __restrict__ w2, const float* __restrict__ b2,
                           float* __restrict__ pos) {
    int bg = blockIdx.x;
    __shared__ float hid[128];
    __shared__ float c3[3];
    int tid = threadIdx.x;
    if (tid < 3) c3[tid] = centers[bg * 3 + tid];
    __syncthreads();
    if (tid < 128) {
        float s = b1[tid];
#pragma unroll
        for (int t = 0; t < 3; t++) s += c3[t] * w1[t * 128 + tid];
        hid[tid] = 0.5f * s * (1.f + erff(s * 0.70710678118654752f));
    }
    __syncthreads();
    float s = b2[tid];
    for (int j = 0; j < 128; j++) s += hid[j] * w2[j * 256 + tid];
    pos[(size_t)bg * 256 + tid] = s;
}

// ---------------- assemble [cls|tokens] and [cls_pos|pos] ----------------
__global__ void build_seq_kernel(const float* __restrict__ tokens, const float* __restrict__ posb,
                                 const float* __restrict__ cls_token, const float* __restrict__ cls_pos,
                                 float* __restrict__ xt, float* __restrict__ ptb) {
    int i = blockIdx.x * 256 + threadIdx.x;  // B*257*256 total
    int c = i & 255;
    int n = (i >> 8) % TT;
    int b = i / (TT * 256);
    if (n == 0) { xt[i] = cls_token[c]; ptb[i] = cls_pos[c]; }
    else {
        int src = (b * GG + n - 1) * 256 + c;
        xt[i] = tokens[src]; ptb[i] = posb[src];
    }
}

// ---------------- add + layernorm (padd / xi_out optional) ----------------
__global__ void add_ln_kernel(const float* __restrict__ xin, const float* __restrict__ padd,
                              const float* __restrict__ g, const float* __restrict__ b,
                              float* __restrict__ xi_out, float* __restrict__ h_out) {
    int row = blockIdx.x, tid = threadIdx.x;
    __shared__ float red[256];
    __shared__ float stat[2];
    float v = xin[row * 256 + tid];
    if (padd) v += padd[row * 256 + tid];
    if (xi_out) xi_out[row * 256 + tid] = v;
    red[tid] = v; __syncthreads();
    for (int s = 128; s > 0; s >>= 1) { if (tid < s) red[tid] += red[tid + s]; __syncthreads(); }
    if (tid == 0) stat[0] = red[0] * (1.f / 256.f);
    __syncthreads();
    float d = v - stat[0];
    red[tid] = d * d; __syncthreads();
    for (int s = 128; s > 0; s >>= 1) { if (tid < s) red[tid] += red[tid + s]; __syncthreads(); }
    if (tid == 0) stat[1] = rsqrtf(red[0] * (1.f / 256.f) + 1e-5f);
    __syncthreads();
    h_out[row * 256 + tid] = d * stat[1] * g[tid] + b[tid];
}

// ---------------- generic WMMA GEMM with TDM double-buffered B slabs ----------------
// C = act(A@W + bias) (+ residual). grid: (N/128, M/16), block 256 (8 waves).
// dyn LDS = 16*K*2 (A, f16) + 2*32*128*2 (B slab double buffer)
// act: 0=none, 1=relu, 2=gelu
__global__ void gemm_wmma_kernel(const float* __restrict__ A, const _Float16* __restrict__ W,
                                 const float* __restrict__ bias, const float* __restrict__ residual,
                                 float* __restrict__ C, int M, int K, int N, int act) {
    extern __shared__ char smem[];
    _Float16* sA = (_Float16*)smem;                        // 16 x K
    _Float16* sB = (_Float16*)(smem + (size_t)16 * K * 2); // 2 x (32 x 128)
    int m0   = blockIdx.y * 16;
    int nblk = blockIdx.x * 128;
    int tid  = threadIdx.x;
    int wave = tid >> 5, lane = tid & 31;

    // stage A row-tile (f32 -> f16) into LDS
    for (int i = tid; i < 16 * K; i += 256) {
        int r = i / K, c = i - r * K;
        sA[i] = (_Float16)A[(size_t)(m0 + r) * K + c];
    }
    // DMA first B slab (rows 0..31, cols nblk..nblk+127) into buffer 0
    stage_b_slab(W, N, K, 0, nblk, sB, wave, tid);
    stage_b_wait(wave);
    __syncthreads();

    int nt = wave;  // n-tile 0..7 within the 128-col block tile
    v8f acc = {0.f,0.f,0.f,0.f,0.f,0.f,0.f,0.f};
    for (int k0 = 0; k0 < K; k0 += 32) {
        int buf = (k0 >> 5) & 1;
        // overlap: DMA next slab into the other buffer while computing this one
        if (k0 + 32 < K) stage_b_slab(W, N, K, k0 + 32, nblk, sB + (buf ^ 1) * 4096, wave, tid);
        v16h a = load_a_lds(sA, K, 0, k0, lane);
        v16h b = load_b_lds(sB + buf * 4096, nt * 16, lane);
        acc = wmma16(a, b, acc);
        stage_b_wait(wave);
        __syncthreads();
    }

    int col = nblk + nt * 16 + (lane & 15);
    int rbase = m0 + ((lane >= 16) ? 8 : 0);
    float bv = bias ? bias[col] : 0.f;
#pragma unroll
    for (int v = 0; v < 8; v++) {
        float val = acc[v] + bv;
        if (act == 1) val = fmaxf(val, 0.f);
        else if (act == 2) val = 0.5f * val * (1.f + erff(val * 0.70710678118654752f));
        size_t o = (size_t)(rbase + v) * N + col;
        if (residual) val += residual[o];
        C[o] = val;
    }
}

// ---------------- attention: one block per (b,h) ----------------
__global__ void attn_kernel(const float* __restrict__ qkv, float* __restrict__ out) {
    int bh = blockIdx.x;
    int b = bh >> 3, h = bh & 7;
    __shared__ float sc[TT];
    __shared__ float red[256];
    __shared__ float qv[DH];
    __shared__ float stat[2];
    int tid = threadIdx.x;
    const float scale = 0.17677669529663687f; // 32^-0.5
    for (int q = 0; q < TT; q++) {
        if (tid < DH) qv[tid] = qkv[((size_t)(b * TT + q)) * 768 + h * DH + tid];
        __syncthreads();
        for (int m = tid; m < TT; m += 256) {
            const float* kp = qkv + ((size_t)(b * TT + m)) * 768 + 256 + h * DH;
            float s = 0.f;
#pragma unroll
            for (int d = 0; d < DH; d++) s += qv[d] * kp[d];
            sc[m] = s * scale;
        }
        __syncthreads();
        float lm = -3.4e38f;
        for (int m = tid; m < TT; m += 256) lm = fmaxf(lm, sc[m]);
        red[tid] = lm; __syncthreads();
        for (int s = 128; s > 0; s >>= 1) { if (tid < s) red[tid] = fmaxf(red[tid], red[tid + s]); __syncthreads(); }
        if (tid == 0) stat[0] = red[0];
        __syncthreads();
        float ls = 0.f;
        for (int m = tid; m < TT; m += 256) { float e = __expf(sc[m] - stat[0]); sc[m] = e; ls += e; }
        red[tid] = ls; __syncthreads();
        for (int s = 128; s > 0; s >>= 1) { if (tid < s) red[tid] += red[tid + s]; __syncthreads(); }
        if (tid == 0) stat[1] = 1.f / red[0];
        __syncthreads();
        if (tid < DH) {
            float acc = 0.f;
            for (int m = 0; m < TT; m++)
                acc += sc[m] * qkv[((size_t)(b * TT + m)) * 768 + 512 + h * DH + tid];
            out[((size_t)(b * TT + q)) * 256 + h * DH + tid] = acc * stat[1];
        }
        __syncthreads();
    }
}

// ---------------- head + scatter: one block (128 thr) per group ----------------
// key optimization: concat's first 256 dims (gfeat) are constant across the 64
// points of a group -> precompute base = gfeat@W1[0:256]+b1 once per group.
__global__ void head_kernel(const float* __restrict__ xt_final, const float* __restrict__ neigh,
                            const int* __restrict__ idx,
                            const float* __restrict__ w1, const float* __restrict__ b1,
                            const float* __restrict__ w2, const float* __restrict__ b2,
                            float* __restrict__ sums, float* __restrict__ cnts) {
    int bg = blockIdx.x;
    int b = bg >> 8, g = bg & 255;
    __shared__ float gf[256];
    __shared__ float base[128];
    __shared__ float red[128];
    int tid = threadIdx.x; // 128
    for (int i = tid; i < 256; i += 128) gf[i] = xt_final[((size_t)(b * TT + 1 + g)) * 256 + i];
    __syncthreads();
    {
        float s = b1[tid];
        for (int c = 0; c < 256; c++) s += gf[c] * w1[c * 128 + tid];
        base[tid] = s;
    }
    __syncthreads();
    for (int kk = 0; kk < KK; kk++) {
        const float* nb = neigh + ((size_t)bg * KK + kk) * 3;
        float s = base[tid];
#pragma unroll
        for (int t = 0; t < 3; t++) s += nb[t] * w1[(256 + t) * 128 + tid];
        s = fmaxf(s, 0.f);
        red[tid] = s * w2[tid];
        __syncthreads();
        for (int st = 64; st > 0; st >>= 1) { if (tid < st) red[tid] += red[tid + st]; __syncthreads(); }
        if (tid == 0) {
            float lg = red[0] + b2[0];
            int j = idx[bg * KK + kk];
            atomicAdd(&sums[(size_t)b * NN + j], lg);
            atomicAdd(&cnts[(size_t)b * NN + j], 1.f);
        }
        __syncthreads();
    }
}

__global__ void finalize_kernel(const float* __restrict__ sums, const float* __restrict__ cnts,
                                float* __restrict__ out, int n) {
    int i = blockIdx.x * 256 + threadIdx.x;
    if (i < n) out[i] = sums[i] / fmaxf(cnts[i], 1.f);
}

// ---------------- host launch ----------------
extern "C" void kernel_launch(void* const* d_in, const int* in_sizes, int n_in,
                              void* d_out, int out_size, void* d_ws, size_t ws_size,
                              hipStream_t stream) {
    (void)in_sizes; (void)n_in; (void)out_size; (void)ws_size;
    const float* x        = (const float*)d_in[0];
    const float* enc_w1   = (const float*)d_in[1];
    const float* enc_b1   = (const float*)d_in[2];
    const float* enc_bn1g = (const float*)d_in[3];
    const float* enc_bn1b = (const float*)d_in[4];
    const float* enc_w2   = (const float*)d_in[5];
    const float* enc_b2   = (const float*)d_in[6];
    const float* enc_w3   = (const float*)d_in[7];
    const float* enc_b3   = (const float*)d_in[8];
    const float* enc_bn2g = (const float*)d_in[9];
    const float* enc_bn2b = (const float*)d_in[10];
    const float* enc_w4   = (const float*)d_in[11];
    const float* enc_b4   = (const float*)d_in[12];
    const float* cls_tok  = (const float*)d_in[13];
    const float* cls_pos  = (const float*)d_in[14];
    const float* pos_w1   = (const float*)d_in[15];
    const float* pos_b1   = (const float*)d_in[16];
    const float* pos_w2   = (const float*)d_in[17];
    const float* pos_b2   = (const float*)d_in[18];
    const float* ln1_g    = (const float*)d_in[19];
    const float* ln1_b    = (const float*)d_in[20];
    const float* qkv_w    = (const float*)d_in[21];
    const float* proj_w   = (const float*)d_in[22];
    const float* proj_b   = (const float*)d_in[23];
    const float* ln2_g    = (const float*)d_in[24];
    const float* ln2_b    = (const float*)d_in[25];
    const float* fc1_w    = (const float*)d_in[26];
    const float* fc1_b    = (const float*)d_in[27];
    const float* fc2_w    = (const float*)d_in[28];
    const float* fc2_b    = (const float*)d_in[29];
    const float* norm_g   = (const float*)d_in[30];
    const float* norm_b   = (const float*)d_in[31];
    const float* head_w1  = (const float*)d_in[32];
    const float* head_b1  = (const float*)d_in[33];
    const float* head_w2  = (const float*)d_in[34];
    const float* head_b2  = (const float*)d_in[35];

    char* ws = (char*)d_ws;
    size_t off = 0;
    auto alloc = [&](size_t bytes) -> char* {
        char* p = ws + off;
        off += (bytes + 255) & ~(size_t)255;
        return p;
    };
    float* centers = (float*)alloc((size_t)BB * GG * 3 * 4);
    int*   idxb    = (int*)  alloc((size_t)BB * GG * KK * 4);
    float* neigh   = (float*)alloc((size_t)BB * GG * KK * 3 * 4);
    float* tokens  = (float*)alloc((size_t)BB * GG * DD * 4);
    float* posb    = (float*)alloc((size_t)BB * GG * DD * 4);
    float* xt      = (float*)alloc((size_t)ROWS * DD * 4);
    float* ptb     = (float*)alloc((size_t)ROWS * DD * 4);
    float* xi      = (float*)alloc((size_t)ROWS * DD * 4);
    float* h1n     = (float*)alloc((size_t)ROWS * DD * 4);
    float* qkvb    = (float*)alloc((size_t)ROWS * 768 * 4);
    float* attno   = (float*)alloc((size_t)ROWS * DD * 4);
    float* h2n     = (float*)alloc((size_t)ROWS * DD * 4);
    float* fcmid   = (float*)alloc((size_t)ROWS * HID * 4);
    float* sums    = (float*)alloc((size_t)BB * NN * 4);
    float* cnts    = (float*)alloc((size_t)BB * NN * 4);
    _Float16* w2h   = (_Float16*)alloc((size_t)128 * 256 * 2);
    _Float16* w3h   = (_Float16*)alloc((size_t)512 * 512 * 2);
    _Float16* w4h   = (_Float16*)alloc((size_t)512 * 256 * 2);
    _Float16* qkvwh = (_Float16*)alloc((size_t)4 * 256 * 768 * 2);
    _Float16* projh = (_Float16*)alloc((size_t)4 * 256 * 256 * 2);
    _Float16* fc1h  = (_Float16*)alloc((size_t)4 * 256 * 1024 * 2);
    _Float16* fc2h  = (_Float16*)alloc((size_t)4 * 1024 * 256 * 2);

    auto cvt = [&](const float* s, _Float16* d, int n) {
        f32_to_f16_kernel<<<(n + 255) / 256, 256, 0, stream>>>(s, d, n);
    };
    cvt(enc_w2, w2h, 128 * 256);
    cvt(enc_w3, w3h, 512 * 512);
    cvt(enc_w4, w4h, 512 * 256);
    cvt(qkv_w, qkvwh, 4 * 256 * 768);
    cvt(proj_w, projh, 4 * 256 * 256);
    cvt(fc1_w, fc1h, 4 * 256 * 1024);
    cvt(fc2_w, fc2h, 4 * 1024 * 256);

    zero_kernel<<<(BB * NN + 255) / 256, 256, 0, stream>>>(sums, BB * NN);
    zero_kernel<<<(BB * NN + 255) / 256, 256, 0, stream>>>(cnts, BB * NN);

    fps_kernel<<<BB, 256, 0, stream>>>(x, centers, idxb);
    knn_kernel<<<BB * GG, 256, 0, stream>>>(x, centers, idxb, neigh);
    encoder_kernel<<<BB * GG, 256, 131072, stream>>>(neigh,
        enc_w1, enc_b1, enc_bn1g, enc_bn1b,
        w2h, enc_b2, w3h, enc_b3, enc_bn2g, enc_bn2b, w4h, enc_b4, tokens);
    pos_kernel<<<BB * GG, 256, 0, stream>>>(centers, pos_w1, pos_b1, pos_w2, pos_b2, posb);
    build_seq_kernel<<<(ROWS * DD) / 256, 256, 0, stream>>>(tokens, posb, cls_tok, cls_pos, xt, ptb);

    const int slabBytes = 2 * 32 * 128 * 2; // B double buffer
    for (int l = 0; l < 4; l++) {
        add_ln_kernel<<<ROWS, 256, 0, stream>>>(xt, ptb, ln1_g + l * DD, ln1_b + l * DD, xi, h1n);
        gemm_wmma_kernel<<<dim3(768 / 128, ROWS / 16), 256, 16 * 256 * 2 + slabBytes, stream>>>(
            h1n, qkvwh + (size_t)l * 256 * 768, nullptr, nullptr, qkvb, ROWS, 256, 768, 0);
        attn_kernel<<<BB * HH, 256, 0, stream>>>(qkvb, attno);
        gemm_wmma_kernel<<<dim3(256 / 128, ROWS / 16), 256, 16 * 256 * 2 + slabBytes, stream>>>(
            attno, projh + (size_t)l * 256 * 256, proj_b + l * DD, xi, xt, ROWS, 256, 256, 0);
        add_ln_kernel<<<ROWS, 256, 0, stream>>>(xt, nullptr, ln2_g + l * DD, ln2_b + l * DD, nullptr, h2n);
        gemm_wmma_kernel<<<dim3(1024 / 128, ROWS / 16), 256, 16 * 256 * 2 + slabBytes, stream>>>(
            h2n, fc1h + (size_t)l * 256 * 1024, fc1_b + l * HID, nullptr, fcmid, ROWS, 256, 1024, 2);
        gemm_wmma_kernel<<<dim3(256 / 128, ROWS / 16), 256, 16 * 1024 * 2 + slabBytes, stream>>>(
            fcmid, fc2h + (size_t)l * 1024 * 256, fc2_b + l * DD, xt, xt, ROWS, 1024, 256, 0);
    }
    add_ln_kernel<<<ROWS, 256, 0, stream>>>(xt, nullptr, norm_g, norm_b, nullptr, h1n);

    head_kernel<<<BB * GG, 128, 0, stream>>>(h1n, neigh, idxb, head_w1, head_b1, head_w2, head_b2,
                                             sums, cnts);
    finalize_kernel<<<(BB * NN + 255) / 256, 256, 0, stream>>>(sums, cnts, (float*)d_out, BB * NN);
}
